// InnerNetFFN_2645699854822
// MI455X (gfx1250) — compile-verified, matrix operand order checked by reading, not run on hardware
//
#include <hip/hip_runtime.h>

typedef __attribute__((ext_vector_type(16))) __bf16 v16bf;
typedef __attribute__((ext_vector_type(8)))  __bf16 v8bf;
typedef __attribute__((ext_vector_type(8)))  float  v8f;

union FragBF { v16bf v; v8bf h[2]; };

__device__ __forceinline__ unsigned pack2bf(float lo, float hi) {
  union { __bf16 b[2]; unsigned u; } u;
  u.b[0] = (__bf16)lo; u.b[1] = (__bf16)hi;
  return u.u;
}

// Async global->LDS 16B copy (gfx1250, GVS mode: SADDR base + 32-bit voffset).
// LDS destination address = low 32 bits of the generic pointer to __shared__.
__device__ __forceinline__ void async_b128(void* ldsPtr, const void* base, unsigned byteOff) {
  const unsigned lds = (unsigned)(size_t)ldsPtr;
  asm volatile("global_load_async_to_lds_b128 %0, %1, %2"
               :: "v"(lds), "v"(byteOff), "s"(base)
               : "memory");
}

template <int N>
__device__ __forceinline__ void wait_async() {
  asm volatile("s_wait_asynccnt %0" :: "i"(N) : "memory");
}

// A/B fragment load for v_wmma_f32_16x16x32_bf16 from an LDS tile stored
// [row][k], ld=32 bf16. Lanes 0-15: row=r, K chunks {0-7,16-23};
// lanes 16-31: row=r, K chunks {8-15,24-31}  (ISA 7.12.2).
__device__ __forceinline__ v16bf load_frag(const __bf16* base, int rowOff, int r, int hb) {
  FragBF f;
  const __bf16* p = base + (rowOff + r) * 32;
  f.h[0] = *reinterpret_cast<const v8bf*>(p + hb * 8);
  f.h[1] = *reinterpret_cast<const v8bf*>(p + 16 + hb * 8);
  return f.v;
}

// ---------------------------------------------------------------------------
// One-time f32 -> bf16 conversion (x, W1a, W1b, W2 staged into workspace).
// ---------------------------------------------------------------------------
__global__ __launch_bounds__(256, 1)
void cvt_f32_to_bf16(const float* __restrict__ src, __bf16* __restrict__ dst, int n4) {
  const int i = blockIdx.x * 256 + threadIdx.x;
  if (i < n4) {
    const float4 v = reinterpret_cast<const float4*>(src)[i];
    uint2 p;
    p.x = pack2bf(v.x, v.y);
    p.y = pack2bf(v.z, v.w);
    reinterpret_cast<uint2*>(dst)[i] = p;
  }
}

// ---------------------------------------------------------------------------
// Stage 1: a = x*W1a^T + b1a ; b = x*W1b^T + b1b ; act = InnerNet(a,b) -> bf16
// Grid: (d_ff/128, M/128), block 256 (8 waves, 2x4), wave tile 64x32.
// Double-buffered async global->LDS pipeline.
// ---------------------------------------------------------------------------
__global__ __launch_bounds__(256, 1)
void innernet_ffn_stage1(const __bf16* __restrict__ xb,
                         const __bf16* __restrict__ w1ab,
                         const __bf16* __restrict__ w1bb,
                         const float* __restrict__ b1a,
                         const float* __restrict__ b1b,
                         const float* __restrict__ Wi1,
                         const float* __restrict__ bi1,
                         const float* __restrict__ Wi2,
                         const float* __restrict__ bi2,
                         __bf16* __restrict__ act)
{
  constexpr int DM  = 1024;       // K (d_model)
  constexpr int DF  = 4096;       // act leading dim (d_ff)
  constexpr int KT  = 32;
  constexpr int NKT = DM / KT;    // 32 K-steps

  __shared__ __bf16 sA [2][128 * KT];
  __shared__ __bf16 sBa[2][128 * KT];
  __shared__ __bf16 sBb[2][128 * KT];
  __shared__ float4 sInner[32];   // {Wi1[h][0], Wi1[h][1], bi1[h], Wi2[h]}

  const int t    = threadIdx.x;
  const int m0   = blockIdx.y * 128;
  const int n0   = blockIdx.x * 128;
  const int wave = t >> 5;
  const int lane = t & 31;
  const int wm   = wave >> 2;     // 0..1 -> 64-row band
  const int wn   = wave & 3;      // 0..3 -> 32-col band
  const int r    = lane & 15;
  const int hb   = lane >> 4;

  const int lrow = t >> 2;        // 64 rows per pass (16B per thread)
  const int lcol = (t & 3) * 8;   // 8 bf16

  if (t < 32) sInner[t] = make_float4(Wi1[2 * t], Wi1[2 * t + 1], bi1[t], Wi2[t]);
  const float cbias = bi2[0];

  v8f accA[8] = {};
  v8f accB[8] = {};

  auto issue = [&](int buf, int k0) {
#pragma unroll
    for (int p = 0; p < 2; ++p) {
      const int row = p * 64 + lrow;
      const unsigned offX = (unsigned)(((m0 + row) * DM + k0 + lcol) * 2);
      const unsigned offW = (unsigned)(((n0 + row) * DM + k0 + lcol) * 2);
      async_b128(&sA [buf][row * KT + lcol], xb,   offX);
      async_b128(&sBa[buf][row * KT + lcol], w1ab, offW);
      async_b128(&sBb[buf][row * KT + lcol], w1bb, offW);
    }
  };

  issue(0, 0);  // prefetch tile 0

  for (int kt = 0; kt < NKT; ++kt) {
    const int cur = kt & 1;
    if (kt + 1 < NKT) {
      issue(cur ^ 1, (kt + 1) * KT);  // overlap next tile with compute
      wait_async<6>();                // drain tile kt (6 older async ops/wave)
    } else {
      wait_async<0>();
    }
    __syncthreads();                  // tile kt visible from all waves

    v16bf aF[4];
#pragma unroll
    for (int sm = 0; sm < 4; ++sm) aF[sm] = load_frag(sA[cur], wm * 64 + sm * 16, r, hb);
    v16bf baF[2], bbF[2];
#pragma unroll
    for (int sn = 0; sn < 2; ++sn) {
      baF[sn] = load_frag(sBa[cur], wn * 32 + sn * 16, r, hb);
      bbF[sn] = load_frag(sBb[cur], wn * 32 + sn * 16, r, hb);
    }
#pragma unroll
    for (int sm = 0; sm < 4; ++sm) {
#pragma unroll
      for (int sn = 0; sn < 2; ++sn) {
        const int i = sm * 2 + sn;
        accA[i] = __builtin_amdgcn_wmma_f32_16x16x32_bf16(
            false, aF[sm], false, baF[sn], (short)0, accA[i], false, false);
        accB[i] = __builtin_amdgcn_wmma_f32_16x16x32_bf16(
            false, aF[sm], false, bbF[sn], (short)0, accB[i], false, false);
      }
    }
    __syncthreads();                  // all reads of buf[cur] done before reuse
  }

  // Epilogue: bias + fused per-element 2->32->1 MLP, write activated as bf16.
  // C/D layout: vgpr rr -> row (hb*8 + rr), lane -> col r (ISA 7.12.2).
#pragma unroll
  for (int sm = 0; sm < 4; ++sm) {
    const int mB = m0 + wm * 64 + sm * 16 + hb * 8;
#pragma unroll
    for (int sn = 0; sn < 2; ++sn) {
      const int i  = sm * 2 + sn;
      const int nG = n0 + wn * 32 + sn * 16 + r;
      const float ba = b1a[nG];
      const float bb = b1b[nG];
#pragma unroll
      for (int rr = 0; rr < 8; ++rr) {
        const float av = accA[i][rr] + ba;
        const float bv = accB[i][rr] + bb;
        float s = cbias;
#pragma unroll 4
        for (int h = 0; h < 32; ++h) {
          const float4 wv = sInner[h];
          const float hv = fmaf(av, wv.x, fmaf(bv, wv.y, wv.z));
          s = fmaf(wv.w, fmaxf(hv, 0.0f), s);
        }
        act[(size_t)(mB + rr) * DF + nG] = (__bf16)s;
      }
    }
  }
}

// ---------------------------------------------------------------------------
// Stage 2: out = act * W2^T + b2   (act, W2 both bf16 in workspace)
// Grid: (1024/128, M/128), block 256, wave tile 64x32, async double-buffered.
// ---------------------------------------------------------------------------
__global__ __launch_bounds__(256, 1)
void innernet_ffn_stage2(const __bf16* __restrict__ act,
                         const __bf16* __restrict__ w2b,
                         const float* __restrict__ b2,
                         float* __restrict__ out)
{
  constexpr int DF  = 4096;     // K
  constexpr int DN  = 1024;     // out leading dim
  constexpr int KT  = 32;
  constexpr int NKT = DF / KT;  // 128 K-steps

  __shared__ __bf16 sA[2][128 * KT];
  __shared__ __bf16 sB[2][128 * KT];

  const int t    = threadIdx.x;
  const int m0   = blockIdx.y * 128;
  const int n0   = blockIdx.x * 128;
  const int wave = t >> 5;
  const int lane = t & 31;
  const int wm   = wave >> 2;
  const int wn   = wave & 3;
  const int r    = lane & 15;
  const int hb   = lane >> 4;

  const int lrow = t >> 2;
  const int lcol = (t & 3) * 8;

  v8f acc[8] = {};

  auto issue = [&](int buf, int k0) {
#pragma unroll
    for (int p = 0; p < 2; ++p) {
      const int row = p * 64 + lrow;
      const unsigned offA = (unsigned)(((m0 + row) * DF + k0 + lcol) * 2);
      const unsigned offB = (unsigned)(((n0 + row) * DF + k0 + lcol) * 2);
      async_b128(&sA[buf][row * KT + lcol], act, offA);
      async_b128(&sB[buf][row * KT + lcol], w2b, offB);
    }
  };

  issue(0, 0);

  for (int kt = 0; kt < NKT; ++kt) {
    const int cur = kt & 1;
    if (kt + 1 < NKT) {
      issue(cur ^ 1, (kt + 1) * KT);
      wait_async<4>();
    } else {
      wait_async<0>();
    }
    __syncthreads();

    v16bf aF[4];
#pragma unroll
    for (int sm = 0; sm < 4; ++sm) aF[sm] = load_frag(sA[cur], wm * 64 + sm * 16, r, hb);
    v16bf bF[2];
#pragma unroll
    for (int sn = 0; sn < 2; ++sn) bF[sn] = load_frag(sB[cur], wn * 32 + sn * 16, r, hb);
#pragma unroll
    for (int sm = 0; sm < 4; ++sm) {
#pragma unroll
      for (int sn = 0; sn < 2; ++sn) {
        const int i = sm * 2 + sn;
        acc[i] = __builtin_amdgcn_wmma_f32_16x16x32_bf16(
            false, aF[sm], false, bF[sn], (short)0, acc[i], false, false);
      }
    }
    __syncthreads();
  }

#pragma unroll
  for (int sm = 0; sm < 4; ++sm) {
    const int mB = m0 + wm * 64 + sm * 16 + hb * 8;
#pragma unroll
    for (int sn = 0; sn < 2; ++sn) {
      const int i  = sm * 2 + sn;
      const int nG = n0 + wn * 32 + sn * 16 + r;
      const float bias = b2[nG];
#pragma unroll
      for (int rr = 0; rr < 8; ++rr) {
        out[(size_t)(mB + rr) * DN + nG] = acc[i][rr] + bias;
      }
    }
  }
}

extern "C" void kernel_launch(void* const* d_in, const int* in_sizes, int n_in,
                              void* d_out, int out_size, void* d_ws, size_t ws_size,
                              hipStream_t stream) {
  const float* x   = (const float*)d_in[0];
  const float* W1a = (const float*)d_in[1];
  const float* b1a = (const float*)d_in[2];
  const float* W1b = (const float*)d_in[3];
  const float* b1b = (const float*)d_in[4];
  const float* Wi1 = (const float*)d_in[5];
  const float* bi1 = (const float*)d_in[6];
  const float* Wi2 = (const float*)d_in[7];
  const float* bi2 = (const float*)d_in[8];
  const float* W2  = (const float*)d_in[9];
  const float* b2  = (const float*)d_in[10];

  // Workspace layout (bf16 elements):
  //   [0,4M)    x_bf16    (4096 x 1024)
  //   [4M,8M)   W1a_bf16  (4096 x 1024)
  //   [8M,12M)  W1b_bf16  (4096 x 1024)
  //   [12M,16M) W2_bf16   (1024 x 4096)
  //   [16M,32M) act_bf16  (4096 x 4096)
  constexpr size_t FOUR_M = 4u * 1024u * 1024u;
  __bf16* ws   = (__bf16*)d_ws;
  __bf16* xb   = ws;
  __bf16* w1ab = ws + 1 * FOUR_M;
  __bf16* w1bb = ws + 2 * FOUR_M;
  __bf16* w2b  = ws + 3 * FOUR_M;
  __bf16* actb = ws + 4 * FOUR_M;

  dim3 blk(256, 1, 1);
  const int n4 = (int)(FOUR_M / 4);          // float4 count per 4M-element array
  const int cvtGrid = n4 / 256;              // 4096 blocks

  cvt_f32_to_bf16<<<cvtGrid, blk, 0, stream>>>(x,   xb,   n4);
  cvt_f32_to_bf16<<<cvtGrid, blk, 0, stream>>>(W1a, w1ab, n4);
  cvt_f32_to_bf16<<<cvtGrid, blk, 0, stream>>>(W1b, w1bb, n4);
  cvt_f32_to_bf16<<<cvtGrid, blk, 0, stream>>>(W2,  w2b,  n4);

  innernet_ffn_stage1<<<dim3(32, 32, 1), blk, 0, stream>>>(
      xb, w1ab, w1bb, b1a, b1b, Wi1, bi1, Wi2, bi2, actb);
  innernet_ffn_stage2<<<dim3(8, 32, 1), blk, 0, stream>>>(
      actb, w2b, b2, (float*)d_out);
}
